// tfidfTree_764504179045
// MI455X (gfx1250) — matrix operand order repaired and checked by reading, not compiled, not run on hardware
//
#include <hip/hip_runtime.h>
#include <hip/hip_bf16.h>
#include <math.h>

// ---------------- problem constants ----------------
#define BB    32
#define LL    128
#define DD    300
#define HH    1024
#define KSTEP 1344           // 1024 (h) + 320 (x padded from 300)
#define NG    4096           // 4*H LSTM gates
#define KCOMP 3072           // 3*H
#define NC    6144           // 6*H compose gates
#define NODE_MAX 127

typedef __bf16 bf16;
typedef __attribute__((ext_vector_type(16))) __bf16 v16bf;
typedef __attribute__((ext_vector_type(8)))  __bf16 v8bf;
typedef __attribute__((ext_vector_type(8)))  float  v8f;

union V16U { v16bf v; v8bf h[2]; };

__device__ __forceinline__ float sigf(float x) { return 1.0f / (1.0f + expf(-x)); }

// ---------------------------------------------------------------------------
// Tree build: one thread per batch item. Emulates the recursive greedy build
// with an explicit stack, emitting compose records in post-order (root last).
// codes: -1 = None/zero state, 0..127 = leaf index, 1000+s = compose slot s.
// ---------------------------------------------------------------------------
__global__ void k_build_tree(const int* __restrict__ words,
                             const int* __restrict__ length,
                             int* __restrict__ n_comp,
                             int* __restrict__ rec_l,
                             int* __restrict__ rec_r,
                             int* __restrict__ rec_x) {
    int b = threadIdx.x;
    if (b >= BB) return;
    const int* w = words + b * LL;
    int len = length[b];
    if (len > LL) len = LL;
    if (len < 0)  len = 0;

    short fs[LL + 2], fe[LL + 2], fpos[LL + 2];
    signed char fstage[LL + 2];
    int flc[LL + 2];

    int sp = 0, n = 0, ret = -1;
    if (len > 0) { fs[0] = 0; fe[0] = (short)len; fstage[0] = 0; sp = 1; }

    while (sp > 0) {
        int i = sp - 1;
        int s = fs[i], e = fe[i];
        if (e - s <= 1) { ret = (e == s) ? -1 : s; sp--; continue; }
        if (fstage[i] == 0) {
            // argmax of 1/(2+df) == first minimum of df = w % 1000
            int bestdf = 0x7fffffff, pos = s;
            for (int t = s; t < e; ++t) {
                int df = (int)(((unsigned)w[t]) % 1000u);
                if (df < bestdf) { bestdf = df; pos = t; }
            }
            fpos[i] = (short)pos; fstage[i] = 1;
            fs[sp] = (short)s; fe[sp] = (short)pos; fstage[sp] = 0; sp++;
            continue;
        }
        if (fstage[i] == 1) {
            flc[i] = ret; fstage[i] = 2;
            fs[sp] = (short)(fpos[i] + 1); fe[sp] = (short)e; fstage[sp] = 0; sp++;
            continue;
        }
        // stage 2: both children done
        rec_l[b * NODE_MAX + n] = flc[i];
        rec_r[b * NODE_MAX + n] = ret;
        rec_x[b * NODE_MAX + n] = fpos[i];
        ret = 1000 + n; n++;
        sp--;
    }
    n_comp[b] = n;
}

// ---------------------------------------------------------------------------
// One-time conversion: fused step weight [4096][1344] bf16, compose weight
// [6144][3072] bf16, padded bf16 embeddings X [B][L][320], and the initial
// step A matrix (h part = 0, x part = x_0).
// ---------------------------------------------------------------------------
__global__ void k_convert(const float* __restrict__ W_ih,
                          const float* __restrict__ W_hh,
                          const float* __restrict__ W_comp,
                          const float* __restrict__ emb,
                          bf16* __restrict__ W_step,
                          bf16* __restrict__ W_comp_bf,
                          bf16* __restrict__ X_bf,
                          bf16* __restrict__ A_step) {
    const size_t n0 = (size_t)NG * KSTEP;
    const size_t n1 = (size_t)NC * KCOMP;
    const size_t n2 = (size_t)BB * LL * 320;
    const size_t n3 = (size_t)BB * KSTEP;
    const size_t total = n0 + n1 + n2 + n3;
    for (size_t idx = (size_t)blockIdx.x * blockDim.x + threadIdx.x; idx < total;
         idx += (size_t)gridDim.x * blockDim.x) {
        if (idx < n0) {
            size_t n = idx / KSTEP, k = idx % KSTEP;
            float v = 0.0f;
            if (k < HH)        v = W_hh[n * HH + k];
            else if (k < HH + DD) v = W_ih[n * DD + (k - HH)];
            W_step[idx] = (bf16)v;
        } else if (idx < n0 + n1) {
            size_t i = idx - n0;
            W_comp_bf[i] = (bf16)W_comp[i];
        } else if (idx < n0 + n1 + n2) {
            size_t i = idx - n0 - n1;
            size_t bt = i / 320, k = i % 320;
            X_bf[i] = (bf16)((k < DD) ? emb[bt * DD + k] : 0.0f);
        } else {
            size_t i = idx - n0 - n1 - n2;
            size_t b = i / KSTEP, k = i % KSTEP;
            bf16 v = (bf16)0.0f;
            if (k >= HH) {
                size_t k2 = k - HH;
                v = (bf16)((k2 < DD) ? emb[(b * LL) * DD + k2] : 0.0f);
            }
            A_step[i] = v;
        }
    }
}

// ---------------------------------------------------------------------------
// GEMM: C[32][N] = A[32][K](bf16) * W[N][K](bf16)^T + bias0[n] (+ bias1[n])
// 256 threads = 8 waves; wave w owns n-tile (blockIdx.x*128 + w*16),
// both m-tiles (rows 0..15 and 16..31). K must be a multiple of 32.
// Operand fragments follow the CDNA5 16-bit WMMA VGPR layouts (ISA 7.12.2).
// ---------------------------------------------------------------------------
__global__ void __launch_bounds__(256)
k_gemm_bf16(const bf16* __restrict__ A, const bf16* __restrict__ W, int K,
            const float* __restrict__ bias0, const float* __restrict__ bias1,
            float* __restrict__ C, int N) {
    const int tid  = threadIdx.x;
    const int wave = tid >> 5;
    const int lane = tid & 31;
    const int half = (lane < 16) ? 0 : 1;        // half-wave
    const int nloc = lane & 15;
    const int n    = blockIdx.x * 128 + wave * 16 + nloc;

    const int kloB = half ? 8  : 0;              // B: lane holds K in {klo..klo+7, 16+klo..16+klo+7}
    const int kloA = half ? 8  : 0;              // A: lane holds K in {klo..klo+7, 16+klo..16+klo+7}

    const bf16* __restrict__ wrow  = W + (size_t)n * K;
    const bf16* __restrict__ arow0 = A + (size_t)nloc * K;          // m-tile 0: rows 0..15
    const bf16* __restrict__ arow1 = A + (size_t)(nloc + 16) * K;   // m-tile 1: rows 16..31

    v8f acc0 = {0.f, 0.f, 0.f, 0.f, 0.f, 0.f, 0.f, 0.f};
    v8f acc1 = {0.f, 0.f, 0.f, 0.f, 0.f, 0.f, 0.f, 0.f};

    for (int k = 0; k < K; k += 32) {
        V16U bfrag, a0, a1;
        bfrag.h[0] = *(const v8bf*)(wrow  + k + kloB);
        bfrag.h[1] = *(const v8bf*)(wrow  + k + 16 + kloB);
        a0.h[0]    = *(const v8bf*)(arow0 + k + kloA);
        a0.h[1]    = *(const v8bf*)(arow0 + k + 16 + kloA);
        a1.h[0]    = *(const v8bf*)(arow1 + k + kloA);
        a1.h[1]    = *(const v8bf*)(arow1 + k + 16 + kloA);
        acc0 = __builtin_amdgcn_wmma_f32_16x16x32_bf16(false, a0.v, false, bfrag.v,
                                                       (short)0, acc0, false, false);
        acc1 = __builtin_amdgcn_wmma_f32_16x16x32_bf16(false, a1.v, false, bfrag.v,
                                                       (short)0, acc1, false, false);
    }

    float bsum = bias0[n] + (bias1 ? bias1[n] : 0.0f);
    const int mtop = half ? 8 : 0;               // C: VGPR g -> row g (or 8+g)
#pragma unroll
    for (int g = 0; g < 8; ++g) {
        C[(size_t)(mtop + g)      * N + n] = acc0[g] + bsum;
        C[(size_t)(16 + mtop + g) * N + n] = acc1[g] + bsum;
    }
}

// ---------------------------------------------------------------------------
// LSTM elementwise: c,h from gates; updates cs, hs_bf, and the step-A matrix
// (h part for step t+1, x part = x_{t+1}).
// ---------------------------------------------------------------------------
__global__ void k_lstm_elem(const float* __restrict__ gates,
                            float* __restrict__ cs,
                            bf16* __restrict__ hs_bf,
                            bf16* __restrict__ A_step,
                            const bf16* __restrict__ X_bf,
                            int t) {
    int idx = blockIdx.x * blockDim.x + threadIdx.x;
    if (idx >= BB * HH) return;
    int b = idx >> 10, j = idx & (HH - 1);
    const float* g = gates + (size_t)b * NG;
    float gi = g[j], gf = g[HH + j], gg = g[2 * HH + j], go = g[3 * HH + j];
    float cp = (t == 0) ? 0.0f : cs[((size_t)b * LL + (t - 1)) * HH + j];
    float c  = sigf(gf) * cp + sigf(gi) * tanhf(gg);
    float h  = sigf(go) * tanhf(c);
    cs[((size_t)b * LL + t) * HH + j]   = c;
    bf16 hb = (bf16)h;
    hs_bf[((size_t)b * LL + t) * HH + j] = hb;
    A_step[(size_t)b * KSTEP + j] = hb;
    if (j < 320 && (t + 1) < LL)
        A_step[(size_t)b * KSTEP + HH + j] = X_bf[((size_t)b * LL + (t + 1)) * 320 + j];
}

// ---------------------------------------------------------------------------
// Compose gather: A_comp[b] = bf16 concat [h_l, h_x, h_r] for record r of
// batch b (zeros if batch b has fewer records).
// ---------------------------------------------------------------------------
__global__ void k_comp_gather(int r,
                              const int* __restrict__ n_comp,
                              const int* __restrict__ rec_l,
                              const int* __restrict__ rec_r,
                              const int* __restrict__ rec_x,
                              const bf16* __restrict__ hs_bf,
                              const float* __restrict__ node_h,
                              bf16* __restrict__ A_comp) {
    int b = blockIdx.x;
    bool active = (r < n_comp[b]);
    int lc = 0, rc = 0, xp = 0;
    if (active) {
        lc = rec_l[b * NODE_MAX + r];
        rc = rec_r[b * NODE_MAX + r];
        xp = rec_x[b * NODE_MAX + r];
    }
    for (int j = threadIdx.x; j < KCOMP; j += blockDim.x) {
        bf16 v = (bf16)0.0f;
        if (active) {
            if (j < HH) {
                int code = lc;
                if (code >= 1000)      v = (bf16)node_h[((size_t)b * NODE_MAX + (code - 1000)) * HH + j];
                else if (code >= 0)    v = hs_bf[((size_t)b * LL + code) * HH + j];
            } else if (j < 2 * HH) {
                int jj = j - HH;
                v = hs_bf[((size_t)b * LL + xp) * HH + jj];
            } else {
                int jj = j - 2 * HH, code = rc;
                if (code >= 1000)      v = (bf16)node_h[((size_t)b * NODE_MAX + (code - 1000)) * HH + jj];
                else if (code >= 0)    v = hs_bf[((size_t)b * LL + code) * HH + jj];
            }
        }
        A_comp[(size_t)b * KCOMP + j] = v;
    }
}

__device__ __forceinline__ float child_c(int code, int b, int j,
                                         const float* cs, const float* node_c) {
    if (code < 0)    return 0.0f;
    if (code < LL)   return cs[((size_t)b * LL + code) * HH + j];
    return node_c[((size_t)b * NODE_MAX + (code - 1000)) * HH + j];
}

// ---------------------------------------------------------------------------
// Compose elementwise: gates (i, fl, fx, fr, u, o) -> node h,c at slot r.
// ---------------------------------------------------------------------------
__global__ void k_comp_elem(int r,
                            const int* __restrict__ n_comp,
                            const int* __restrict__ rec_l,
                            const int* __restrict__ rec_r,
                            const int* __restrict__ rec_x,
                            const float* __restrict__ comp_gates,
                            const float* __restrict__ cs,
                            float* __restrict__ node_h,
                            float* __restrict__ node_c) {
    int b = blockIdx.x;
    if (r >= n_comp[b]) return;
    int lc = rec_l[b * NODE_MAX + r];
    int rc = rec_r[b * NODE_MAX + r];
    int xp = rec_x[b * NODE_MAX + r];
    const float* g = comp_gates + (size_t)b * NC;
    for (int j = threadIdx.x; j < HH; j += blockDim.x) {
        float gi = g[j],            gfl = g[HH + j],     gfx = g[2 * HH + j];
        float gfr = g[3 * HH + j],  gu  = g[4 * HH + j], go  = g[5 * HH + j];
        float cl = child_c(lc, b, j, cs, node_c);
        float cx = cs[((size_t)b * LL + xp) * HH + j];
        float cr = child_c(rc, b, j, cs, node_c);
        float c = sigf(gi) * tanhf(gu) + sigf(gfl) * cl + sigf(gfx) * cx + sigf(gfr) * cr;
        float h = sigf(go) * tanhf(c);
        node_h[((size_t)b * NODE_MAX + r) * HH + j] = h;
        node_c[((size_t)b * NODE_MAX + r) * HH + j] = c;
    }
}

// ---------------------------------------------------------------------------
// Output: root = last post-order record (or leaf 0 if single-token sentence).
// d_out = [h (32x1024), c (32x1024)].
// ---------------------------------------------------------------------------
__global__ void k_output(const int* __restrict__ n_comp,
                         const int* __restrict__ length,
                         const float* __restrict__ node_h,
                         const float* __restrict__ node_c,
                         const bf16* __restrict__ hs_bf,
                         const float* __restrict__ cs,
                         float* __restrict__ out) {
    int b = blockIdx.x;
    int nb = n_comp[b];
    int len = length[b];
    for (int j = threadIdx.x; j < HH; j += blockDim.x) {
        float h = 0.0f, c = 0.0f;
        if (nb > 0) {
            h = node_h[((size_t)b * NODE_MAX + (nb - 1)) * HH + j];
            c = node_c[((size_t)b * NODE_MAX + (nb - 1)) * HH + j];
        } else if (len >= 1) {
            h = (float)hs_bf[(size_t)b * LL * HH + j];
            c = cs[(size_t)b * LL * HH + j];
        }
        out[(size_t)b * HH + j]             = h;
        out[(size_t)BB * HH + (size_t)b * HH + j] = c;
    }
}

// ---------------------------------------------------------------------------
extern "C" void kernel_launch(void* const* d_in, const int* in_sizes, int n_in,
                              void* d_out, int out_size, void* d_ws, size_t ws_size,
                              hipStream_t stream) {
    const float* emb    = (const float*)d_in[0];   // (B,L,D)
    const float* W_ih   = (const float*)d_in[1];   // (4H,D)
    const float* W_hh   = (const float*)d_in[2];   // (4H,H)
    const float* b_ih   = (const float*)d_in[3];   // (4H)
    const float* b_hh   = (const float*)d_in[4];   // (4H)
    const float* W_comp = (const float*)d_in[5];   // (6H,3H)
    const float* b_comp = (const float*)d_in[6];   // (6H)
    const int*   words  = (const int*)d_in[7];     // (B,L)
    const int*   length = (const int*)d_in[8];     // (B)
    float* out = (float*)d_out;

    // carve workspace (256B aligned slabs)
    char* p = (char*)d_ws;
    auto carve = [&](size_t bytes) -> void* {
        void* r = (void*)p;
        p += (bytes + 255) & ~(size_t)255;
        return r;
    };
    bf16*  W_step_bf = (bf16*) carve((size_t)NG * KSTEP * 2);
    bf16*  W_comp_bf = (bf16*) carve((size_t)NC * KCOMP * 2);
    bf16*  X_bf      = (bf16*) carve((size_t)BB * LL * 320 * 2);
    bf16*  A_step    = (bf16*) carve((size_t)BB * KSTEP * 2);
    float* gates_tmp = (float*)carve((size_t)BB * NG * 4);
    float* cs        = (float*)carve((size_t)BB * LL * HH * 4);
    bf16*  hs_bf     = (bf16*) carve((size_t)BB * LL * HH * 2);
    bf16*  A_comp    = (bf16*) carve((size_t)BB * KCOMP * 2);
    float* comp_gate = (float*)carve((size_t)BB * NC * 4);
    float* node_h    = (float*)carve((size_t)BB * NODE_MAX * HH * 4);
    float* node_c    = (float*)carve((size_t)BB * NODE_MAX * HH * 4);
    int*   n_comp    = (int*)  carve(BB * 4);
    int*   rec_l     = (int*)  carve(BB * NODE_MAX * 4);
    int*   rec_r     = (int*)  carve(BB * NODE_MAX * 4);
    int*   rec_x     = (int*)  carve(BB * NODE_MAX * 4);

    // 1. tree structure (scalar, tiny)
    k_build_tree<<<1, 32, 0, stream>>>(words, length, n_comp, rec_l, rec_r, rec_x);

    // 2. bf16 weight/embedding conversion + initial step-A
    k_convert<<<4096, 256, 0, stream>>>(W_ih, W_hh, W_comp, emb,
                                        W_step_bf, W_comp_bf, X_bf, A_step);

    // 3. leaf LSTM: 128 sequential steps of [32x1344]@[1344x4096] + pointwise
    for (int t = 0; t < LL; ++t) {
        k_gemm_bf16<<<NG / 128, 256, 0, stream>>>(A_step, W_step_bf, KSTEP,
                                                  b_ih, b_hh, gates_tmp, NG);
        k_lstm_elem<<<(BB * HH) / 256, 256, 0, stream>>>(gates_tmp, cs, hs_bf,
                                                         A_step, X_bf, t);
    }

    // 4. compose: up to 127 rounds, each a batched [32x3072]@[3072x6144]
    for (int r = 0; r < NODE_MAX; ++r) {
        k_comp_gather<<<BB, 256, 0, stream>>>(r, n_comp, rec_l, rec_r, rec_x,
                                              hs_bf, node_h, A_comp);
        k_gemm_bf16<<<NC / 128, 256, 0, stream>>>(A_comp, W_comp_bf, KCOMP,
                                                  b_comp, (const float*)nullptr,
                                                  comp_gate, NC);
        k_comp_elem<<<BB, 256, 0, stream>>>(r, n_comp, rec_l, rec_r, rec_x,
                                            comp_gate, cs, node_h, node_c);
    }

    // 5. root states -> d_out
    k_output<<<BB, 256, 0, stream>>>(n_comp, length, node_h, node_c, hs_bf, cs, out);
}
